// DoubleHiddenTransitionModel_58884001628272
// MI455X (gfx1250) — matrix operand order repaired and checked by reading, not compiled, not run on hardware
//
#include <hip/hip_runtime.h>

// ---------------------------------------------------------------------------
// Problem dims (fixed by the reference)
// ---------------------------------------------------------------------------
#define B_    1024
#define S_    1024
#define D_    1024
#define E_    1024
#define H_    1024
#define T_    32
#define KQ    3072         // 2S + E       (posterior embed K)
#define KP_IN 2072         // 2S + 6*4     (prior embed K, raw)
#define KP    2112         // padded to a multiple of 64 (24 pose + 40 zero + 2S)
#define MIN_STD 0.1f

#define LDS_ROW   144      // 64 K * 2B = 128B data + 16B pad (bank-conflict-free)
#define LDS_STAGE (64 * LDS_ROW)

typedef __attribute__((ext_vector_type(16))) __bf16 v16bf;
typedef __attribute__((ext_vector_type(8)))  float  v8f;

// ---------------------------------------------------------------------------
// Weight conversion: f32 -> bf16 (weights stay L2-resident: ~48MB << 192MB)
// ---------------------------------------------------------------------------
__global__ void cvt_f32_bf16(const float* __restrict__ in, __bf16* __restrict__ out, int n) {
    int i = blockIdx.x * 256 + threadIdx.x;
    if (i < n) out[i] = (__bf16)in[i];
}

// W_esp (1024 x 2072) -> padded (1024 x 2112): cols [24,64) are zero.
__global__ void cvt_wesp(const float* __restrict__ in, __bf16* __restrict__ out) {
    int i = blockIdx.x * 256 + threadIdx.x;           // over 1024*2112
    if (i >= D_ * KP) return;
    int n = i / KP, k = i % KP;
    float v;
    if (k < 24)      v = in[n * KP_IN + k];
    else if (k < 64) v = 0.0f;
    else             v = in[n * KP_IN + (k - 40)];
    out[i] = (__bf16)v;
}

// ---------------------------------------------------------------------------
// Concat/pack kernels (f32 activations)
// ---------------------------------------------------------------------------
__global__ void pack_xq(const float* __restrict__ obs, const float* __restrict__ sq,
                        const float* __restrict__ sp, float* __restrict__ x) {
    int i = blockIdx.x * 256 + threadIdx.x;           // over B*KQ
    int b = i / KQ, k = i % KQ;
    float v;
    if (k < E_)           v = obs[b * E_ + k];
    else if (k < E_ + S_) v = sq[b * S_ + (k - E_)];
    else                  v = sp[b * S_ + (k - E_ - S_)];
    x[i] = v;
}

__global__ void pack_xp(const float* __restrict__ pose, const float* __restrict__ sq,
                        const float* __restrict__ sp, float* __restrict__ x) {
    int i = blockIdx.x * 256 + threadIdx.x;           // over B*KP
    int b = i / KP, k = i % KP;
    float v;
    if (k < 24)           v = pose[b * 6 + (k % 6)];  // jnp.tile(pose,(1,4))
    else if (k < 64)      v = 0.0f;                   // pad (weights also zero here)
    else if (k < 64 + S_) v = sq[b * S_ + (k - 64)];
    else                  v = sp[b * S_ + (k - 64 - S_)];
    x[i] = v;
}

// ---------------------------------------------------------------------------
// WMMA GEMM:  C[M=1024,N] = act(A[1024,K](f32) @ W[N,K](bf16)^T + bias[N])
// Block: 128 threads = 4 waves; block tile 128(M) x 64(N); each wave: 32x64
// (two A fragments x four B fragments -> 8 WMMAs per K=32 half, so each
// ds-loaded B fragment feeds TWO v_wmma ops -> 2x arithmetic intensity).
// B tile (64N x 64K bf16, 8KB) is DMA'd to LDS with global_load_async_to_lds,
// double-buffered (2 stages in flight, ASYNCcnt-gated). 16 WMMAs/stage/wave.
// No divergence on EXEC anywhere (WMMA requires EXEC all-ones).
// ---------------------------------------------------------------------------
template <int ACT>   // 0 = identity, 1 = relu
__global__ __launch_bounds__(128)
void gemm_wmma_bf16(const float* __restrict__ A, const __bf16* __restrict__ W,
                    const float* __restrict__ bias, float* __restrict__ C,
                    int K, int N) {
    __shared__ __attribute__((aligned(16))) unsigned char ldsraw[2 * LDS_STAGE];

    const int tid  = threadIdx.x;
    const int lane = tid & 31;
    const int wid  = tid >> 5;                      // 0..3
    const int hi   = lane >> 4;
    const int lo   = lane & 15;
    const int rowBase = blockIdx.y * 128 + wid * 32;
    const int colBase = blockIdx.x * 64;

    // --- async B-tile staging: thread t copies 64B: row = t/2, 64B-half = t%2 ---
    const int srow = tid >> 1;                      // 0..63  (tile column = weight row)
    const int shalf = tid & 1;                      // 0..1
    const unsigned ldsbase = (unsigned)(size_t)ldsraw;
    const unsigned lslot   = ldsbase + srow * LDS_ROW + shalf * 64;
    const __bf16*  gbase   = W + (size_t)(colBase + srow) * K + shalf * 32;

    auto issue_stage = [&](int slot, int kb) {
        unsigned loff = lslot + (unsigned)slot * LDS_STAGE;
        unsigned long long ga = (unsigned long long)(gbase + kb);
        asm volatile("global_load_async_to_lds_b128 %0, %1, off"
                     :: "v"(loff), "v"(ga) : "memory");
        asm volatile("global_load_async_to_lds_b128 %0, %1, off offset:16"
                     :: "v"(loff), "v"(ga) : "memory");
        asm volatile("global_load_async_to_lds_b128 %0, %1, off offset:32"
                     :: "v"(loff), "v"(ga) : "memory");
        asm volatile("global_load_async_to_lds_b128 %0, %1, off offset:48"
                     :: "v"(loff), "v"(ga) : "memory");
    };

    v8f acc[2][4];
    #pragma unroll
    for (int nf = 0; nf < 4; ++nf) {
        float bv = bias[colBase + nf * 16 + lo];
        #pragma unroll
        for (int r = 0; r < 8; ++r) { acc[0][nf][r] = bv; acc[1][nf][r] = bv; }
    }

    const float* arow0 = A + (size_t)(rowBase + lo) * K;        // M rows  0..15 of tile
    const float* arow1 = arow0 + (size_t)16 * K;                // M rows 16..31 of tile

    const int nstage = K >> 6;                      // K is a multiple of 64, >= 1024
    issue_stage(0, 0);
    issue_stage(1, 64);

    for (int s = 0; s < nstage; ++s) {
        const int kb = s << 6;
        // Async loads complete in order: with two stages (4 ops each) in flight,
        // cnt<=4 completes the current stage; on the last stage drain fully.
        if (s + 1 < nstage) asm volatile("s_wait_asynccnt 0x4" ::: "memory");
        else                asm volatile("s_wait_asynccnt 0x0" ::: "memory");
        __syncthreads();                            // LDS stage visible to all waves

        const unsigned char* buf = ldsraw + (s & 1) * LDS_STAGE;

        #pragma unroll
        for (int kh = 0; kh < 2; ++kh) {
            // A fragments (16x32 bf16 each): lane<16 holds K = {0..15} of
            // its row; lane>=16 holds K = {16..31}.  f32 load + cvt.
            const int kg = kb + kh * 32 + (hi << 4);
            v16bf a[2];
            #pragma unroll
            for (int mh = 0; mh < 2; ++mh) {
                const float* ar = (mh == 0) ? arow0 : arow1;
                float4 f0 = *(const float4*)(ar + kg);
                float4 f1 = *(const float4*)(ar + kg + 4);
                float4 f2 = *(const float4*)(ar + kg + 8);
                float4 f3 = *(const float4*)(ar + kg + 12);
                v16bf t;
                t[0]=(__bf16)f0.x;  t[1]=(__bf16)f0.y;  t[2]=(__bf16)f0.z;  t[3]=(__bf16)f0.w;
                t[4]=(__bf16)f1.x;  t[5]=(__bf16)f1.y;  t[6]=(__bf16)f1.z;  t[7]=(__bf16)f1.w;
                t[8]=(__bf16)f2.x;  t[9]=(__bf16)f2.y;  t[10]=(__bf16)f2.z; t[11]=(__bf16)f2.w;
                t[12]=(__bf16)f3.x; t[13]=(__bf16)f3.y; t[14]=(__bf16)f3.z; t[15]=(__bf16)f3.w;
                a[mh] = t;
            }

            // B fragments: lane holds 16 contiguous K of column colBase+nf*16+lo
            // (padded 144B LDS stride -> conflict-free).  Each fragment feeds
            // two WMMAs (both M halves) -> ds traffic halved per WMMA.
            #pragma unroll
            for (int nf = 0; nf < 4; ++nf) {
                const unsigned char* bp =
                    buf + (nf * 16 + lo) * LDS_ROW + (kh * 32 + hi * 16) * 2;
                union { uint4 u[2]; v16bf v; } t;
                t.u[0] = *(const uint4*)bp;
                t.u[1] = *(const uint4*)(bp + 16);
                acc[0][nf] = __builtin_amdgcn_wmma_f32_16x16x32_bf16(
                    false, a[0], false, t.v, (short)0, acc[0][nf], false, false);
                acc[1][nf] = __builtin_amdgcn_wmma_f32_16x16x32_bf16(
                    false, a[1], false, t.v, (short)0, acc[1][nf], false, false);
            }
        }

        __syncthreads();                            // all reads done before refill
        if (s + 2 < nstage) issue_stage(s & 1, kb + 128);
    }

    // C/D layout: VGPR r holds M = rowBase + mh*16 + r + 8*hi, N = colBase+nf*16+lo
    #pragma unroll
    for (int mh = 0; mh < 2; ++mh) {
        #pragma unroll
        for (int nf = 0; nf < 4; ++nf) {
            const int n = colBase + nf * 16 + lo;
            #pragma unroll
            for (int r = 0; r < 8; ++r) {
                const int m = rowBase + mh * 16 + hi * 8 + r;
                float v = acc[mh][nf][r];
                if (ACT) v = fmaxf(v, 0.0f);
                C[(size_t)m * N + n] = v;
            }
        }
    }
}

// ---------------------------------------------------------------------------
// GRU gate combine: gi/gh are (B, 3D) with [r | z | n] splits, biases already
// folded in by the GEMMs.  In-place safe (hnew may alias hold elementwise).
// ---------------------------------------------------------------------------
__global__ void gru_combine(const float* __restrict__ gi, const float* __restrict__ gh,
                            const float* __restrict__ hold, float* __restrict__ hnew) {
    int i = blockIdx.x * 256 + threadIdx.x;           // over B*D
    int b = i >> 10, d = i & (D_ - 1);
    const float* gib = gi + (size_t)b * 3 * D_;
    const float* ghb = gh + (size_t)b * 3 * D_;
    float r = 1.0f / (1.0f + expf(-(gib[d] + ghb[d])));
    float z = 1.0f / (1.0f + expf(-(gib[D_ + d] + ghb[D_ + d])));
    float n = tanhf(gib[2 * D_ + d] + r * ghb[2 * D_ + d]);
    hnew[i] = (1.0f - z) * n + z * hold[i];
}

// ---------------------------------------------------------------------------
// Head: split (B,2S) -> mean / softplus(raw)+MIN_STD, reparameterize with noise.
// ---------------------------------------------------------------------------
__global__ void head_kernel(const float* __restrict__ o, const float* __restrict__ noise,
                            float* __restrict__ mean, float* __restrict__ stdv,
                            float* __restrict__ state) {
    int i = blockIdx.x * 256 + threadIdx.x;           // over B*S
    int b = i >> 10, s = i & (S_ - 1);
    float m   = o[(size_t)b * 2 * S_ + s];
    float raw = o[(size_t)b * 2 * S_ + S_ + s];
    float sp  = (raw > 20.0f) ? raw : log1pf(expf(raw));
    float st  = sp + MIN_STD;
    mean[i]  = m;
    stdv[i]  = st;
    state[i] = m + st * noise[i];
}

// ---------------------------------------------------------------------------
// Host orchestration: 32 sequential steps, 16 kernels each, all on `stream`.
// ---------------------------------------------------------------------------
extern "C" void kernel_launch(void* const* d_in, const int* in_sizes, int n_in,
                              void* d_out, int out_size, void* d_ws, size_t ws_size,
                              hipStream_t stream) {
    (void)in_sizes; (void)n_in; (void)out_size; (void)ws_size;
    const float* prev_state   = (const float*)d_in[0];
    const float* prev_belief  = (const float*)d_in[1];
    const float* observations = (const float*)d_in[2];
    const float* poses        = (const float*)d_in[3];
    const float* noise_post   = (const float*)d_in[4];
    const float* noise_prior  = (const float*)d_in[5];
    const float* W_esq = (const float*)d_in[6];
    const float* b_esq = (const float*)d_in[7];
    const float* W_esp = (const float*)d_in[8];
    const float* b_esp = (const float*)d_in[9];
    const float* gq_wih = (const float*)d_in[10];
    const float* gq_whh = (const float*)d_in[11];
    const float* gq_bih = (const float*)d_in[12];
    const float* gq_bhh = (const float*)d_in[13];
    const float* gp_wih = (const float*)d_in[14];
    const float* gp_whh = (const float*)d_in[15];
    const float* gp_bih = (const float*)d_in[16];
    const float* gp_bhh = (const float*)d_in[17];
    const float* W_ebq = (const float*)d_in[18];
    const float* b_ebq = (const float*)d_in[19];
    const float* W_sq  = (const float*)d_in[20];
    const float* b_sq  = (const float*)d_in[21];
    const float* W_ebp = (const float*)d_in[22];
    const float* b_ebp = (const float*)d_in[23];
    const float* W_sp  = (const float*)d_in[24];
    const float* b_sp  = (const float*)d_in[25];

    float* out = (float*)d_out;
    const size_t TBS = (size_t)T_ * B_ * S_;
    float* out_post_beliefs = out + 0 * TBS;
    float* out_prior_states = out + 1 * TBS;
    float* out_prior_means  = out + 2 * TBS;
    float* out_prior_stds   = out + 3 * TBS;
    float* out_post_states  = out + 4 * TBS;
    float* out_post_means   = out + 5 * TBS;
    float* out_post_stds    = out + 6 * TBS;

    // --- workspace carve (256B aligned) ---
    char* ws = (char*)d_ws;
    size_t off = 0;
    auto carve = [&](size_t bytes) -> char* {
        char* p = ws + off;
        off += (bytes + 255) & ~(size_t)255;
        return p;
    };
    __bf16* W_esq_bf  = (__bf16*)carve((size_t)D_ * KQ * 2);
    __bf16* W_esp_bf  = (__bf16*)carve((size_t)D_ * KP * 2);
    __bf16* gq_wih_bf = (__bf16*)carve((size_t)3 * D_ * D_ * 2);
    __bf16* gq_whh_bf = (__bf16*)carve((size_t)3 * D_ * D_ * 2);
    __bf16* gp_wih_bf = (__bf16*)carve((size_t)3 * D_ * D_ * 2);
    __bf16* gp_whh_bf = (__bf16*)carve((size_t)3 * D_ * D_ * 2);
    __bf16* W_ebq_bf  = (__bf16*)carve((size_t)H_ * D_ * 2);
    __bf16* W_ebp_bf  = (__bf16*)carve((size_t)H_ * D_ * 2);
    __bf16* W_sq_bf   = (__bf16*)carve((size_t)2 * S_ * H_ * 2);
    __bf16* W_sp_bf   = (__bf16*)carve((size_t)2 * S_ * H_ * 2);
    float* Xq   = (float*)carve((size_t)B_ * KQ * 4);
    float* Xp   = (float*)carve((size_t)B_ * KP * 4);
    float* hbuf = (float*)carve((size_t)B_ * D_ * 4);
    float* gi   = (float*)carve((size_t)B_ * 3 * D_ * 4);
    float* gh   = (float*)carve((size_t)B_ * 3 * D_ * 4);
    float* h2   = (float*)carve((size_t)B_ * H_ * 4);
    float* obuf = (float*)carve((size_t)B_ * 2 * S_ * 4);
    float* bp   = (float*)carve((size_t)B_ * D_ * 4);

    // --- bf16 weight conversion (every call; deterministic) ---
    auto cvt = [&](const float* src, __bf16* dst, size_t n) {
        cvt_f32_bf16<<<(unsigned)((n + 255) / 256), 256, 0, stream>>>(src, dst, (int)n);
    };
    cvt(W_esq, W_esq_bf, (size_t)D_ * KQ);
    cvt_wesp<<<(unsigned)(((size_t)D_ * KP + 255) / 256), 256, 0, stream>>>(W_esp, W_esp_bf);
    cvt(gq_wih, gq_wih_bf, (size_t)3 * D_ * D_);
    cvt(gq_whh, gq_whh_bf, (size_t)3 * D_ * D_);
    cvt(gp_wih, gp_wih_bf, (size_t)3 * D_ * D_);
    cvt(gp_whh, gp_whh_bf, (size_t)3 * D_ * D_);
    cvt(W_ebq, W_ebq_bf, (size_t)H_ * D_);
    cvt(W_ebp, W_ebp_bf, (size_t)H_ * D_);
    cvt(W_sq,  W_sq_bf,  (size_t)2 * S_ * H_);
    cvt(W_sp,  W_sp_bf,  (size_t)2 * S_ * H_);

    // prior-belief carry init
    hipMemcpyAsync(bp, prev_belief, (size_t)B_ * D_ * 4, hipMemcpyDeviceToDevice, stream);

    const float* sq = prev_state;
    const float* sp = prev_state;
    const float* bq = prev_belief;

    auto gemm = [&](const float* A, const __bf16* W, const float* bias, float* C,
                    int K, int N, int relu) {
        dim3 grid(N / 64, B_ / 128);
        if (relu) gemm_wmma_bf16<1><<<grid, 128, 0, stream>>>(A, W, bias, C, K, N);
        else      gemm_wmma_bf16<0><<<grid, 128, 0, stream>>>(A, W, bias, C, K, N);
    };

    const int BD = B_ * D_;
    for (int t = 0; t < T_; ++t) {
        // pack concat inputs from *previous* states (prior uses old s_q/s_p too)
        pack_xq<<<(B_ * KQ) / 256, 256, 0, stream>>>(observations + (size_t)t * B_ * E_, sq, sp, Xq);
        pack_xp<<<(B_ * KP) / 256, 256, 0, stream>>>(poses + (size_t)t * B_ * 6, sq, sp, Xp);

        // ---- posterior branch ----
        gemm(Xq, W_esq_bf, b_esq, hbuf, KQ, D_, 1);
        gemm(hbuf, gq_wih_bf, gq_bih, gi, D_, 3 * D_, 0);
        gemm(bq,   gq_whh_bf, gq_bhh, gh, D_, 3 * D_, 0);
        float* bq_new = out_post_beliefs + (size_t)t * BD;
        gru_combine<<<BD / 256, 256, 0, stream>>>(gi, gh, bq, bq_new);
        gemm(bq_new, W_ebq_bf, b_ebq, h2, D_, H_, 1);
        gemm(h2, W_sq_bf, b_sq, obuf, H_, 2 * S_, 0);
        head_kernel<<<(B_ * S_) / 256, 256, 0, stream>>>(
            obuf, noise_post + (size_t)t * B_ * S_,
            out_post_means + (size_t)t * BD, out_post_stds + (size_t)t * BD,
            out_post_states + (size_t)t * BD);

        // ---- prior branch ----
        gemm(Xp, W_esp_bf, b_esp, hbuf, KP, D_, 1);
        gemm(hbuf, gp_wih_bf, gp_bih, gi, D_, 3 * D_, 0);
        gemm(bp,   gp_whh_bf, gp_bhh, gh, D_, 3 * D_, 0);
        gru_combine<<<BD / 256, 256, 0, stream>>>(gi, gh, bp, bp);   // in-place carry
        gemm(bp, W_ebp_bf, b_ebp, h2, D_, H_, 1);
        gemm(h2, W_sp_bf, b_sp, obuf, H_, 2 * S_, 0);
        head_kernel<<<(B_ * S_) / 256, 256, 0, stream>>>(
            obuf, noise_prior + (size_t)t * B_ * S_,
            out_prior_means + (size_t)t * BD, out_prior_stds + (size_t)t * BD,
            out_prior_states + (size_t)t * BD);

        // advance carries (all live in the output buffer except bp)
        sq = out_post_states + (size_t)t * BD;
        sp = out_prior_states + (size_t)t * BD;
        bq = bq_new;
    }
}
